// VQ_61005715472447
// MI455X (gfx1250) — matrix operand order repaired and checked by reading, not compiled, not run on hardware
//
#include <hip/hip_runtime.h>

typedef __attribute__((ext_vector_type(16))) _Float16 v16h;
typedef __attribute__((ext_vector_type(8)))  float    v8f;

namespace {
constexpr int KC            = 512;             // codebook entries
constexpr int DIM           = 64;              // embedding dim
constexpr int THREADS       = 256;             // 8 waves (wave32)
constexpr int ROWS_PER_WAVE = 16;
constexpr int ROWS_PER_BLOCK = (THREADS / 32) * ROWS_PER_WAVE;   // 128
}

// Zero the single loss accumulator slot (d_out is poisoned with 0xAA).
__global__ void vq_init_loss(float* __restrict__ out, long long n) {
  out[n * (long long)DIM + n] = 0.0f;
}

__global__ __launch_bounds__(THREADS)
void vq_kernel(const float* __restrict__ z,
               const float* __restrict__ cb,
               float* __restrict__ out,
               long long n) {
  // f16 codebook (64 KB) + f32 squared norms (2 KB) in LDS
  __shared__ __align__(32) _Float16 s_cb[KC * DIM];
  __shared__ float s_cn[KC];

  const int tid = threadIdx.x;
  for (int i = tid; i < KC * DIM; i += THREADS)
    s_cb[i] = (_Float16)cb[i];
  for (int k = tid; k < KC; k += THREADS) {
    float s = 0.0f;
#pragma unroll
    for (int d = 0; d < DIM; ++d) { const float c = cb[k * DIM + d]; s = fmaf(c, c, s); }
    s_cn[k] = s;
  }
  __syncthreads();

  const int lane = tid & 31;
  const int wave = tid >> 5;
  const int half = lane >> 4;        // which 16-lane half of the wave
  const int l16  = lane & 15;
  const long long rowbase = (long long)blockIdx.x * ROWS_PER_BLOCK
                          + (long long)wave * ROWS_PER_WAVE;

  // ---- A fragments: 16x32 f16 tiles of z (rows rowbase..rowbase+15) ----
  // Contiguous k order (this code's choice): lane L holds row M=L%16; halves {0..15}->K=o..o+15
  // with o = (L<16 ? 0 : 16).
  v16h a0, a1;
  {
    const float* zr = z + (rowbase + l16) * DIM;
    const int o = half * 16;
#pragma unroll
    for (int j = 0; j < 8; ++j) {
      a0[j]     = (_Float16)zr[o + j];
      a0[8 + j] = (_Float16)zr[o + 8 + j];
      a1[j]     = (_Float16)zr[32 + o + j];
      a1[8 + j] = (_Float16)zr[32 + o + 8 + j];
    }
  }

  // Running argmin of score[k] = ||c_k||^2 - 2 * (z . c_k)   (||z||^2 dropped)
  float best[8];
  int   bidx[8];
#pragma unroll
  for (int v = 0; v < 8; ++v) { best[v] = 3.0e38f; bidx[v] = 0; }

  // Two column tiles per iteration: two independent WMMA accumulator chains
  // give the scheduler work to fill the WMMA->VALU hazard window (no v_nops).
  for (int t = 0; t < KC / 16; t += 2) {
    const int col0 = t * 16 + l16;
    const int col1 = col0 + 16;
    // B fragments: 32x16 f16; lane L holds column N=L%16,
    // halves j -> K = (L<16 ? 0 : 16) + j  (contiguous 32B in LDS).
    const _Float16* cbc = s_cb + col0 * DIM + half * 16;
    const v16h b00 = *(const v16h*)(cbc);
    const v16h b01 = *(const v16h*)(cbc + 32);
    const v16h b10 = *(const v16h*)(cbc + 16 * DIM);
    const v16h b11 = *(const v16h*)(cbc + 16 * DIM + 32);

    v8f c0 = {}, c1 = {};
    c0 = __builtin_amdgcn_wmma_f32_16x16x32_f16(false, a0, false, b00, (short)0, c0, false, false);
    c0 = __builtin_amdgcn_wmma_f32_16x16x32_f16(false, a1, false, b01, (short)0, c0, false, false);
    c1 = __builtin_amdgcn_wmma_f32_16x16x32_f16(false, a0, false, b10, (short)0, c1, false, false);
    c1 = __builtin_amdgcn_wmma_f32_16x16x32_f16(false, a1, false, b11, (short)0, c1, false, false);

    const float cn0 = s_cn[col0];
    const float cn1 = s_cn[col1];
#pragma unroll
    for (int v = 0; v < 8; ++v) {
      const float s0 = fmaf(-2.0f, c0[v], cn0);
      const bool t0 = s0 < best[v];
      best[v] = t0 ? s0 : best[v];
      bidx[v] = t0 ? col0 : bidx[v];
    }
#pragma unroll
    for (int v = 0; v < 8; ++v) {
      const float s1 = fmaf(-2.0f, c1[v], cn1);
      const bool t1 = s1 < best[v];
      best[v] = t1 ? s1 : best[v];
      bidx[v] = t1 ? col1 : bidx[v];
    }
  }

  // Min-reduce across the 16 lanes of each half (wave32 shuffles);
  // branchless tie-break to the lowest codebook index (matches argmin).
#pragma unroll
  for (int v = 0; v < 8; ++v) {
#pragma unroll
    for (int m = 8; m >= 1; m >>= 1) {
      const float ob = __shfl_xor(best[v], m, 32);
      const int   oi = __shfl_xor(bidx[v], m, 32);
      const bool take = (ob < best[v]) | ((ob == best[v]) & (oi < bidx[v]));
      best[v] = take ? ob : best[v];
      bidx[v] = take ? oi : bidx[v];
    }
  }

  // idx16[r] = winning code for row rowbase+r, identical on all 32 lanes.
  int idx16[16];
#pragma unroll
  for (int v = 0; v < 8; ++v) {
    const int oth = __shfl_xor(bidx[v], 16, 32);
    idx16[v]     = half ? oth     : bidx[v];   // rows 0..7  (C-layout M=v,   lanes 0-15)
    idx16[8 + v] = half ? bidx[v] : oth;       // rows 8..15 (C-layout M=v+8, lanes 16-31)
  }

  // ---- epilogue: gather z_q (f32), loss partials, idx ----
  const long long idx_off  = n * (long long)DIM;
  const long long loss_off = idx_off + n;
  float lsum = 0.0f;
#pragma unroll
  for (int r = 0; r < 16; ++r) {
    const int q = idx16[r];                              // uniform across wave
    const long long zrow = (rowbase + r) * DIM;
    const float2 cq = *(const float2*)(cb + (long long)q * DIM + 2 * lane);
    const float2 zv = *(const float2*)(z + zrow + 2 * lane);
    const float d0 = zv.x - cq.x;
    const float d1 = zv.y - cq.y;
    lsum = fmaf(d0, d0, lsum);
    lsum = fmaf(d1, d1, lsum);
    *(float2*)(out + zrow + 2 * lane) = cq;              // z_q_st == z_q numerically
    if (lane == r) out[idx_off + rowbase + r] = (float)q;
  }

  // Wave-reduce loss, one f32 atomic per wave into the pre-zeroed slot.
#pragma unroll
  for (int m = 16; m >= 1; m >>= 1) lsum += __shfl_xor(lsum, m, 32);
  if (lane == 0) {
    const float scale = 2.0f / ((float)n * (float)DIM);  // two identical mean terms
    unsafeAtomicAdd(out + loss_off, lsum * scale);
  }
}

extern "C" void kernel_launch(void* const* d_in, const int* in_sizes, int n_in,
                              void* d_out, int out_size, void* d_ws, size_t ws_size,
                              hipStream_t stream) {
  (void)n_in; (void)out_size; (void)d_ws; (void)ws_size;
  const float* z  = (const float*)d_in[0];
  const float* cb = (const float*)d_in[1];
  float* out = (float*)d_out;
  const long long n = (long long)in_sizes[0] / DIM;      // 524288 rows

  vq_init_loss<<<1, 1, 0, stream>>>(out, n);
  const int blocks = (int)(n / ROWS_PER_BLOCK);          // 4096
  vq_kernel<<<blocks, THREADS, 0, stream>>>(z, cb, out, n);
}
